// DTPViT_5901285064949
// MI455X (gfx1250) — matrix operand order repaired and checked by reading, compiled, not run on hardware
//
#include <hip/hip_runtime.h>
#include <hip/hip_bf16.h>
#include <math.h>

// ---------------- static model config ----------------
#define DMODEL 768
#define H3     2304      // 3*D
#define NHEAD  12
#define DH     64
#define BATCH  64
#define LSEQ   196
#define LP_R   208       // padded attention rows (mult of 16)
#define LP_C   224       // padded attention cols (mult of 32, K dim of attn@V)
#define MLPD   3072
#define NCLS   1000
#define TPRE   (LSEQ*BATCH)        // 12544
#define TSHORT ((LSEQ+1)*BATCH)    // 12608
#define BM 4             // M-tiles per wave in GEMM
#define BN 2             // N-tiles per wave in GEMM

typedef __attribute__((ext_vector_type(16))) _Float16 v16h;
typedef __attribute__((ext_vector_type(8)))  float    v8f;
typedef __attribute__((ext_vector_type(4)))  unsigned int v4u;

union Frag16 { v4u u[2]; v16h h; };

__device__ __forceinline__ float geluf(float x) {
    return 0.5f * x * (1.0f + erff(x * 0.70710678118654752f));
}

// ---------------- utility kernels ----------------
__global__ void k_cvt_f16(const float* __restrict__ in, _Float16* __restrict__ out, size_t n) {
    size_t i = (size_t)blockIdx.x * blockDim.x + threadIdx.x;
    if (i < n) out[i] = (_Float16)in[i];
}

__global__ void k_fill_i32(int* __restrict__ p, int v, size_t n) {
    size_t i = (size_t)blockIdx.x * blockDim.x + threadIdx.x;
    if (i < n) p[i] = v;
}

// im2col: x (64,3,224,224) -> patches f16 in token order t = l*64+b, row = 768 (c,py,px)
__global__ void k_im2col(const float* __restrict__ x, _Float16* __restrict__ out) {
    size_t i = (size_t)blockIdx.x * blockDim.x + threadIdx.x;
    if (i >= (size_t)TPRE * DMODEL) return;
    int t = (int)(i / DMODEL), e = (int)(i % DMODEL);
    int b = t & 63, l = t >> 6;
    int gy = l / 14, gx = l % 14;
    int c = e >> 8, r = e & 255, py = r >> 4, px = r & 15;
    out[i] = (_Float16)x[(((size_t)b * 3 + c) * 224 + gy * 16 + py) * 224 + gx * 16 + px];
}

// ---- WMMA GEMM, register-blocked BMxBN tiles/wave:
// C[M,N] = A[M,K] @ W[N,K]^T (+bias +pos +residual, optional GELU)
__global__ void k_gemm(const _Float16* __restrict__ A, int lda,
                       const _Float16* __restrict__ W, int ldw,
                       const float* __restrict__ bias,
                       const float* __restrict__ residual,
                       const float* __restrict__ posTbl,
                       float* __restrict__ Cf32, _Float16* __restrict__ Cf16,
                       int M, int N, int K, int act) {
    int wv = threadIdx.x >> 5, lane = threadIdx.x & 31;
    int tilesM = (M + 15) >> 4, tilesN = (N + 15) >> 4;
    int gM = (tilesM + BM - 1) / BM, gN = (tilesN + BN - 1) / BN;
    int tile = blockIdx.x * (blockDim.x >> 5) + wv;
    if (tile >= gM * gN) return;
    int tm = tile / gN, tn = tile - tm * gN;
    int g = lane >> 4, ln = lane & 15;

    const _Float16* Ap[BM];
    const _Float16* Wp[BN];
#pragma unroll
    for (int i = 0; i < BM; i++) {
        int ar = (tm * BM + i) * 16 + ln; if (ar >= M) ar = M - 1;
        Ap[i] = A + (size_t)ar * lda;
    }
#pragma unroll
    for (int j = 0; j < BN; j++) {
        int wr = (tn * BN + j) * 16 + ln; if (wr >= N) wr = N - 1;
        Wp[j] = W + (size_t)wr * ldw;
    }

    v8f acc[BM][BN] = {};
    for (int kb = 0; kb < K; kb += 32) {
        Frag16 a[BM], bb[BN];
#pragma unroll
        for (int i = 0; i < BM; i++) {
            a[i].u[0] = *(const v4u*)(Ap[i] + kb + g * 8);
            a[i].u[1] = *(const v4u*)(Ap[i] + kb + 16 + g * 8);
        }
#pragma unroll
        for (int j = 0; j < BN; j++) {
            bb[j].u[0] = *(const v4u*)(Wp[j] + kb + g * 8);
            bb[j].u[1] = *(const v4u*)(Wp[j] + kb + 16 + g * 8);
        }
#pragma unroll
        for (int i = 0; i < BM; i++)
#pragma unroll
            for (int j = 0; j < BN; j++)
                acc[i][j] = __builtin_amdgcn_wmma_f32_16x16x32_f16(
                    false, a[i].h, false, bb[j].h, (short)0, acc[i][j], false, false);
    }

#pragma unroll
    for (int j = 0; j < BN; j++) {
        int col = (tn * BN + j) * 16 + ln;
        if (col >= N) continue;
        float bv = bias ? bias[col] : 0.0f;
#pragma unroll
        for (int i = 0; i < BM; i++) {
#pragma unroll
            for (int r = 0; r < 8; r++) {
                int row = (tm * BM + i) * 16 + r + 8 * g;
                if (row >= M) continue;
                float v = acc[i][j][r] + bv;
                if (posTbl)   v += posTbl[(size_t)(row >> 6) * N + col];
                if (residual) v += residual[(size_t)row * N + col];
                if (act)      v = geluf(v);
                if (Cf32) Cf32[(size_t)row * N + col] = v;
                if (Cf16) Cf16[(size_t)row * N + col] = (_Float16)v;
            }
        }
    }
}

// ---------------- layernorm: f32 (T,768) -> f16 (T,768) ----------------
__global__ void k_ln(const float* __restrict__ h, const float* __restrict__ gw,
                     const float* __restrict__ bw, _Float16* __restrict__ out) {
    int t = blockIdx.x, tid = threadIdx.x;
    const float* row = h + (size_t)t * DMODEL;
    __shared__ float red[256];
    float v0 = row[tid], v1 = row[tid + 256], v2 = row[tid + 512];
    red[tid] = v0 + v1 + v2; __syncthreads();
    for (int s = 128; s > 0; s >>= 1) { if (tid < s) red[tid] += red[tid + s]; __syncthreads(); }
    float mean = red[0] * (1.0f / DMODEL); __syncthreads();
    float d0 = v0 - mean, d1 = v1 - mean, d2 = v2 - mean;
    red[tid] = d0 * d0 + d1 * d1 + d2 * d2; __syncthreads();
    for (int s = 128; s > 0; s >>= 1) { if (tid < s) red[tid] += red[tid + s]; __syncthreads(); }
    float rs = rsqrtf(red[0] * (1.0f / DMODEL) + 1e-5f);
    _Float16* o = out + (size_t)t * DMODEL;
    o[tid]       = (_Float16)(d0 * rs * gw[tid]       + bw[tid]);
    o[tid + 256] = (_Float16)(d1 * rs * gw[tid + 256] + bw[tid + 256]);
    o[tid + 512] = (_Float16)(d2 * rs * gw[tid + 512] + bw[tid + 512]);
}

// ---- attention scores: wave owns one q row-tile, sweeps all 13 k-tiles (A loaded once)
__global__ void k_scores(const _Float16* __restrict__ qkv, _Float16* __restrict__ attn, int Ls) {
    int wv = threadIdx.x >> 5, lane = threadIdx.x & 31;
    int tq = blockIdx.x * 8 + wv;
    if (tq >= 13) return;
    int bh = blockIdx.y, b = bh / NHEAD, hh = bh % NHEAD;
    int g = lane >> 4, ln = lane & 15;
    int l = tq * 16 + ln; int lc = (l < Ls) ? l : Ls - 1;
    const _Float16* qp = qkv + (size_t)(lc * BATCH + b) * H3 + hh * DH;
    Frag16 a0, a1;
    a0.u[0] = *(const v4u*)(qp + g * 8);
    a0.u[1] = *(const v4u*)(qp + 16 + g * 8);
    a1.u[0] = *(const v4u*)(qp + 32 + g * 8);
    a1.u[1] = *(const v4u*)(qp + 48 + g * 8);
    _Float16* outp = attn + (size_t)bh * LP_R * LP_C;
    for (int tk = 0; tk < 13; tk++) {
        int m = tk * 16 + ln; int mc = (m < Ls) ? m : Ls - 1;
        const _Float16* kp = qkv + (size_t)(mc * BATCH + b) * H3 + DMODEL + hh * DH;
        Frag16 b0, b1;
        b0.u[0] = *(const v4u*)(kp + g * 8);
        b0.u[1] = *(const v4u*)(kp + 16 + g * 8);
        b1.u[0] = *(const v4u*)(kp + 32 + g * 8);
        b1.u[1] = *(const v4u*)(kp + 48 + g * 8);
        v8f acc = {};
        acc = __builtin_amdgcn_wmma_f32_16x16x32_f16(false, a0.h, false, b0.h, (short)0, acc, false, false);
        acc = __builtin_amdgcn_wmma_f32_16x16x32_f16(false, a1.h, false, b1.h, (short)0, acc, false, false);
        int col = tk * 16 + ln;
        if (col < Ls) {
#pragma unroll
            for (int r = 0; r < 8; r++) {
                int row = tq * 16 + r + 8 * g;
                if (row < Ls) outp[(size_t)row * LP_C + col] = (_Float16)(acc[r] * 0.125f);
            }
        }
    }
}

__global__ void k_softmax(_Float16* __restrict__ attn, int Ls) {
    int idx = blockIdx.x, bh = idx / LP_R, l = idx % LP_R;
    int tid = threadIdx.x;
    _Float16* rowp = attn + ((size_t)bh * LP_R + l) * LP_C;
    if (l >= Ls) { if (tid < LP_C) rowp[tid] = (_Float16)0.0f; return; }
    __shared__ float red[256];
    float x = (tid < Ls) ? (float)rowp[tid] : -1e30f;
    red[tid] = x; __syncthreads();
    for (int s = 128; s > 0; s >>= 1) { if (tid < s) red[tid] = fmaxf(red[tid], red[tid + s]); __syncthreads(); }
    float mx = red[0]; __syncthreads();
    float e = (tid < Ls) ? __expf(x - mx) : 0.0f;
    red[tid] = e; __syncthreads();
    for (int s = 128; s > 0; s >>= 1) { if (tid < s) red[tid] += red[tid + s]; __syncthreads(); }
    float inv = 1.0f / red[0];
    if (tid < LP_C) rowp[tid] = (tid < Ls) ? (_Float16)(e * inv) : (_Float16)0.0f;
}

// ---- transpose V into (bh, d, m) with zero pad to LP_C, so attn@V uses b128 frag loads
__global__ void k_vtrans(const _Float16* __restrict__ qkv, _Float16* __restrict__ vT, int Ls) {
    size_t i = (size_t)blockIdx.x * blockDim.x + threadIdx.x;
    if (i >= (size_t)BATCH * NHEAD * DH * LP_C) return;
    int m = (int)(i % LP_C);
    int rest = (int)(i / LP_C);
    int d = rest & 63, bh = rest >> 6;
    int b = bh / NHEAD, hh = bh % NHEAD;
    _Float16 v = (_Float16)0.0f;
    if (m < Ls) v = qkv[(size_t)(m * BATCH + b) * H3 + 2 * DMODEL + hh * DH + d];
    vT[i] = v;
}

// ---- o = attn @ V: wave owns one l-tile, A frag reused across 4 d-tiles
__global__ void k_av(const _Float16* __restrict__ attn, const _Float16* __restrict__ vT,
                     _Float16* __restrict__ o16, int Ls) {
    int wv = threadIdx.x >> 5, lane = threadIdx.x & 31;
    int tl = blockIdx.x * 8 + wv;
    if (tl >= 13) return;
    int bh = blockIdx.y, b = bh / NHEAD, hh = bh % NHEAD;
    int g = lane >> 4, ln = lane & 15;
    int l = tl * 16 + ln; int lc = (l < Ls) ? l : Ls - 1;
    const _Float16* ap = attn + ((size_t)bh * LP_R + lc) * LP_C;
    const _Float16* vbase = vT + (size_t)bh * DH * LP_C;
    v8f acc[4] = {};
    for (int kb = 0; kb < LP_C; kb += 32) {
        Frag16 a;
        a.u[0] = *(const v4u*)(ap + kb + g * 8);
        a.u[1] = *(const v4u*)(ap + kb + 16 + g * 8);
#pragma unroll
        for (int td = 0; td < 4; td++) {
            int d = td * 16 + ln;
            const _Float16* vp = vbase + (size_t)d * LP_C;
            Frag16 bb;
            bb.u[0] = *(const v4u*)(vp + kb + g * 8);
            bb.u[1] = *(const v4u*)(vp + kb + 16 + g * 8);
            acc[td] = __builtin_amdgcn_wmma_f32_16x16x32_f16(
                false, a.h, false, bb.h, (short)0, acc[td], false, false);
        }
    }
#pragma unroll
    for (int td = 0; td < 4; td++) {
        int d = td * 16 + ln;
#pragma unroll
        for (int r = 0; r < 8; r++) {
            int row = tl * 16 + r + 8 * g;
            if (row < Ls)
                o16[(size_t)(row * BATCH + b) * DMODEL + hh * DH + d] = (_Float16)acc[td][r];
        }
    }
}

// ---------------- boundary predictor tail ----------------
__global__ void k_bplogit(const _Float16* __restrict__ hid, const float* __restrict__ w2,
                          const float* __restrict__ b2, float* __restrict__ logits) {
    int t = blockIdx.x, tid = threadIdx.x;
    __shared__ float red[256];
    const _Float16* row = hid + (size_t)t * 1536;
    float s = 0.0f;
    for (int i = tid; i < 1536; i += 256) s += (float)row[i] * w2[i];
    red[tid] = s; __syncthreads();
    for (int st = 128; st > 0; st >>= 1) { if (tid < st) red[tid] += red[tid + st]; __syncthreads(); }
    if (tid == 0) { int b = t & 63, l = t >> 6; logits[b * LSEQ + l] = red[0] + b2[0]; }
}

__device__ __forceinline__ float rnd01(unsigned x) {
    x ^= x >> 17; x *= 0xed5ad4bbu; x ^= x >> 11; x *= 0xac4c1b51u;
    x ^= x >> 15; x *= 0x31848babu; x ^= x >> 14;
    float u = (float)(x >> 8) * (1.0f / 16777216.0f);
    return fminf(fmaxf(u, 1e-6f), 1.0f - 1e-6f);
}

// serial per-batch cumsum / segmentation (seg non-decreasing -> contiguous segments)
__global__ void k_boundary(const float* __restrict__ logits, int* __restrict__ cs,
                           int* __restrict__ cnt, int* __restrict__ start) {
    int b = threadIdx.x;
    if (b >= BATCH) return;
    int c = 0;
    for (int l = 0; l < LSEQ; l++) {
        float lg = logits[b * LSEQ + l];
        float u = rnd01((unsigned)(b * LSEQ + l) * 2654435761u + 12345u);
        float z = (lg + logf(u) - log1pf(-u)) * 2.0f;   // / TEMP(0.5)
        float soft = 1.0f / (1.0f + expf(-z));
        int hard = (soft > 0.5f) ? 1 : 0;
        c += hard;
        cs[b * LSEQ + l] = c;
        cnt[b * 197 + (c - hard)] += 1;
    }
    int acc = 0;
    for (int s = 0; s < 197; s++) { start[b * 197 + s] = acc; acc += cnt[b * 197 + s]; }
}

// short tokens: row 0 = null_group, rows 1..196 = mean of h over segment (deterministic gather)
__global__ void k_shortgather(const float* __restrict__ hpre, const int* __restrict__ cnt,
                              const int* __restrict__ start, const float* __restrict__ nullg,
                              float* __restrict__ h2) {
    size_t i = (size_t)blockIdx.x * blockDim.x + threadIdx.x;
    if (i >= (size_t)TSHORT * DMODEL) return;
    int d = (int)(i % DMODEL); int t2 = (int)(i / DMODEL);
    int sp = t2 >> 6, b = t2 & 63;
    float v;
    if (sp == 0) v = nullg[d];
    else {
        int s = sp - 1;
        int c = cnt[b * 197 + s], st = start[b * 197 + s];
        float sum = 0.0f;
        for (int k = 0; k < c; k++) sum += hpre[(size_t)(((st + k) << 6) | b) * DMODEL + d];
        v = sum / ((float)c + 1e-9f);
    }
    h2[i] = v;
}

__global__ void k_upsample(float* __restrict__ hpre, const float* __restrict__ h2,
                           const int* __restrict__ cs) {
    size_t i = (size_t)blockIdx.x * blockDim.x + threadIdx.x;
    if (i >= (size_t)TPRE * DMODEL) return;
    int d = (int)(i % DMODEL); int t = (int)(i / DMODEL);
    int b = t & 63, l = t >> 6;
    int s = cs[b * LSEQ + l];
    hpre[i] += h2[(size_t)((s << 6) | b) * DMODEL + d];
}

__global__ void k_pool(const _Float16* __restrict__ ln16, _Float16* __restrict__ pooled) {
    int b = blockIdx.x, tid = threadIdx.x;
    for (int d = tid; d < DMODEL; d += 256) {
        float s = 0.0f;
        for (int l = 0; l < LSEQ; l++) s += (float)ln16[(size_t)(l * BATCH + b) * DMODEL + d];
        pooled[b * DMODEL + d] = (_Float16)(s * (1.0f / (float)LSEQ));
    }
}

// ---------------- host driver ----------------
extern "C" void kernel_launch(void* const* d_in, const int* in_sizes, int n_in,
                              void* d_out, int out_size, void* d_ws, size_t ws_size,
                              hipStream_t stream) {
    (void)in_sizes; (void)out_size;
    if (!d_in || !d_out || !d_ws || n_in < 157) return;

    const float* x       = (const float*)d_in[0];
    const float* patch_w = (const float*)d_in[1];
    const float* patch_b = (const float*)d_in[2];
    const float* pos     = (const float*)d_in[3];
    struct Blk { const float *ln1g,*ln1b,*Wqkv,*bqkv,*Wo,*bo,*ln2g,*ln2b,*W1,*b1,*W2,*b2; } blk[12];
    for (int j = 0; j < 12; j++) {
        int base = 4 + 12 * j;
        blk[j].ln1g = (const float*)d_in[base + 0];
        blk[j].ln1b = (const float*)d_in[base + 1];
        blk[j].Wqkv = (const float*)d_in[base + 2];
        blk[j].bqkv = (const float*)d_in[base + 3];
        blk[j].Wo   = (const float*)d_in[base + 4];
        blk[j].bo   = (const float*)d_in[base + 5];
        blk[j].ln2g = (const float*)d_in[base + 6];
        blk[j].ln2b = (const float*)d_in[base + 7];
        blk[j].W1   = (const float*)d_in[base + 8];
        blk[j].b1   = (const float*)d_in[base + 9];
        blk[j].W2   = (const float*)d_in[base + 10];
        blk[j].b2   = (const float*)d_in[base + 11];
    }
    const float* bpW1  = (const float*)d_in[148];
    const float* bpb1  = (const float*)d_in[149];
    const float* bpW2  = (const float*)d_in[150];
    const float* bpb2  = (const float*)d_in[151];
    const float* normg = (const float*)d_in[152];
    const float* normb = (const float*)d_in[153];
    const float* headw = (const float*)d_in[154];
    const float* headb = (const float*)d_in[155];
    const float* nullg = (const float*)d_in[156];

    // bump allocator on workspace
    char* p = (char*)d_ws;
    auto alloc = [&](size_t bytes) -> char* {
        char* r = p; p += (bytes + 255) & ~(size_t)255; return r;
    };
    float*     hbuf   = (float*)alloc((size_t)TSHORT * DMODEL * 4);
    float*     hpre   = (float*)alloc((size_t)TPRE   * DMODEL * 4);
    _Float16*  xn16   = (_Float16*)alloc((size_t)TSHORT * DMODEL * 2);
    _Float16*  qkv16  = (_Float16*)alloc((size_t)TSHORT * H3 * 2);
    _Float16*  attn16 = (_Float16*)alloc((size_t)BATCH * NHEAD * LP_R * LP_C * 2);
    _Float16*  vT16   = (_Float16*)alloc((size_t)BATCH * NHEAD * DH * LP_C * 2);
    _Float16*  o16    = (_Float16*)alloc((size_t)TSHORT * DMODEL * 2);
    _Float16*  mlp16  = (_Float16*)alloc((size_t)TSHORT * MLPD * 2);
    _Float16*  patch16= (_Float16*)alloc((size_t)TPRE * DMODEL * 2);
    _Float16*  pooled16 = (_Float16*)alloc((size_t)BATCH * DMODEL * 2);
    float*     logits = (float*)alloc((size_t)BATCH * LSEQ * 4);
    int*       cs     = (int*)alloc((size_t)BATCH * LSEQ * 4);
    int*       cnt    = (int*)alloc((size_t)BATCH * 197 * 4);
    int*       start  = (int*)alloc((size_t)BATCH * 197 * 4);
    _Float16 *wq16[12], *wo16[12], *w116[12], *w216[12];
    for (int j = 0; j < 12; j++) {
        wq16[j] = (_Float16*)alloc((size_t)H3 * DMODEL * 2);
        wo16[j] = (_Float16*)alloc((size_t)DMODEL * DMODEL * 2);
        w116[j] = (_Float16*)alloc((size_t)MLPD * DMODEL * 2);
        w216[j] = (_Float16*)alloc((size_t)DMODEL * MLPD * 2);
    }
    _Float16* patchW16 = (_Float16*)alloc((size_t)DMODEL * DMODEL * 2);
    _Float16* bpW116   = (_Float16*)alloc((size_t)1536 * DMODEL * 2);
    _Float16* headW16  = (_Float16*)alloc((size_t)NCLS * DMODEL * 2);
    if ((size_t)(p - (char*)d_ws) > ws_size) return;   // workspace too small

    auto cvt = [&](const float* src, _Float16* dst, size_t n) {
        k_cvt_f16<<<(unsigned)((n + 255) / 256), 256, 0, stream>>>(src, dst, n);
    };
    auto gemm = [&](const _Float16* A, int lda, const _Float16* W, int ldw,
                    const float* bias, const float* res, const float* ptbl,
                    float* c32, _Float16* c16, int M, int N, int K, int act) {
        int tilesM = (M + 15) >> 4, tilesN = (N + 15) >> 4;
        int gM = (tilesM + BM - 1) / BM, gN = (tilesN + BN - 1) / BN;
        int waves = gM * gN;
        k_gemm<<<(waves + 7) / 8, 256, 0, stream>>>(A, lda, W, ldw, bias, res, ptbl,
                                                    c32, c16, M, N, K, act);
    };
    auto run_block = [&](float* h, int T, int Ls, int j) {
        k_ln<<<T, 256, 0, stream>>>(h, blk[j].ln1g, blk[j].ln1b, xn16);
        gemm(xn16, DMODEL, wq16[j], DMODEL, blk[j].bqkv, nullptr, nullptr,
             nullptr, qkv16, T, H3, DMODEL, 0);
        dim3 gs((13 + 7) / 8, BATCH * NHEAD);
        k_scores<<<gs, 256, 0, stream>>>(qkv16, attn16, Ls);
        k_softmax<<<BATCH * NHEAD * LP_R, 256, 0, stream>>>(attn16, Ls);
        size_t nvt = (size_t)BATCH * NHEAD * DH * LP_C;
        k_vtrans<<<(unsigned)((nvt + 255) / 256), 256, 0, stream>>>(qkv16, vT16, Ls);
        k_av<<<gs, 256, 0, stream>>>(attn16, vT16, o16, Ls);
        gemm(o16, DMODEL, wo16[j], DMODEL, blk[j].bo, h, nullptr,
             h, nullptr, T, DMODEL, DMODEL, 0);
        k_ln<<<T, 256, 0, stream>>>(h, blk[j].ln2g, blk[j].ln2b, xn16);
        gemm(xn16, DMODEL, w116[j], DMODEL, blk[j].b1, nullptr, nullptr,
             nullptr, mlp16, T, MLPD, DMODEL, 1);
        gemm(mlp16, MLPD, w216[j], MLPD, blk[j].b2, h, nullptr,
             h, nullptr, T, DMODEL, MLPD, 0);
    };

    // --- weight conversion to f16 ---
    for (int j = 0; j < 12; j++) {
        cvt(blk[j].Wqkv, wq16[j], (size_t)H3 * DMODEL);
        cvt(blk[j].Wo,   wo16[j], (size_t)DMODEL * DMODEL);
        cvt(blk[j].W1,   w116[j], (size_t)MLPD * DMODEL);
        cvt(blk[j].W2,   w216[j], (size_t)DMODEL * MLPD);
    }
    cvt(patch_w, patchW16, (size_t)DMODEL * DMODEL);
    cvt(bpW1,    bpW116,   (size_t)1536 * DMODEL);
    cvt(headw,   headW16,  (size_t)NCLS * DMODEL);

    // --- patch embedding (+bias +pos_embed) ---
    k_im2col<<<(unsigned)(((size_t)TPRE * DMODEL + 255) / 256), 256, 0, stream>>>(x, patch16);
    gemm(patch16, DMODEL, patchW16, DMODEL, patch_b, nullptr, pos,
         hbuf, nullptr, TPRE, DMODEL, DMODEL, 0);

    // --- pre blocks ---
    run_block(hbuf, TPRE, LSEQ, 0);
    run_block(hbuf, TPRE, LSEQ, 1);

    // --- boundary predictor (uses raw h, no LN) ---
    k_cvt_f16<<<(unsigned)(((size_t)TPRE * DMODEL + 255) / 256), 256, 0, stream>>>(hbuf, xn16, (size_t)TPRE * DMODEL);
    gemm(xn16, DMODEL, bpW116, DMODEL, bpb1, nullptr, nullptr,
         nullptr, mlp16, TPRE, 1536, DMODEL, 1);
    k_bplogit<<<TPRE, 256, 0, stream>>>(mlp16, bpW2, bpb2, logits);
    k_fill_i32<<<(BATCH * 197 + 255) / 256, 256, 0, stream>>>(cnt, 0, (size_t)BATCH * 197);
    k_boundary<<<1, 64, 0, stream>>>(logits, cs, cnt, start);

    // --- downsample into short tokens ---
    hipMemcpyAsync(hpre, hbuf, (size_t)TPRE * DMODEL * 4, hipMemcpyDeviceToDevice, stream);
    k_shortgather<<<(unsigned)(((size_t)TSHORT * DMODEL + 255) / 256), 256, 0, stream>>>(
        hpre, cnt, start, nullg, hbuf);

    // --- short blocks (197 tokens) ---
    for (int j = 2; j < 10; j++) run_block(hbuf, TSHORT, LSEQ + 1, j);

    // --- upsample + residual ---
    k_upsample<<<(unsigned)(((size_t)TPRE * DMODEL + 255) / 256), 256, 0, stream>>>(hpre, hbuf, cs);

    // --- post blocks ---
    run_block(hpre, TPRE, LSEQ, 10);
    run_block(hpre, TPRE, LSEQ, 11);

    // --- final LN, mean pool, head ---
    k_ln<<<TPRE, 256, 0, stream>>>(hpre, normg, normb, xn16);
    k_pool<<<BATCH, 256, 0, stream>>>(xn16, pooled16);
    gemm(pooled16, DMODEL, headW16, DMODEL, headb, nullptr, nullptr,
         (float*)d_out, nullptr, BATCH, NCLS, DMODEL, 0);
}
